// MoE_LoRA_83992380440979
// MI455X (gfx1250) — matrix operand order repaired and checked by reading, not compiled, not run on hardware
//
#include <hip/hip_runtime.h>

// ---------------------------------------------------------------------------
// MoE-LoRA attention forward for MI455X (gfx1250, wave32, WMMA bf16).
//   B=8, S=1024, T=8192, DIM=768, H=12, HD=64, E=8, R=16, TOPK=2, SCALING=2.
// All heavy GEMMs on V_WMMA_F32_16X16X32_BF16 (f32 accumulate).
// B operands pre-packed as K-pair dwords so every fragment load is b128 and
// every global->LDS staging is a straight async copy (double-buffered GEMM).
// ---------------------------------------------------------------------------

#define T_TOK   8192
#define DIM     768
#define QKV3    2304
#define NHEAD   12
#define HD      64
#define NEXP    8
#define RANK    16
#define ER      128          // NEXP*RANK
#define KCAT    928          // DIM + ER + NEXP + 24 pad  (multiple of 32)
#define SCALING 2.0f
#define ATT_SCALE 0.125f     // 64^-0.5 (exact power of two -> fold into Q)

#define A_STRIDE 40          // padded LDS row stride for A tiles (ushorts)
#define B_STRIDE 20          // padded LDS row stride for B tiles (uints)
#define K_STRIDE 36          // padded LDS row stride for K tiles (uints)
#define V_STRIDE 20          // padded LDS row stride for V tiles (uints)

typedef __attribute__((ext_vector_type(16))) __bf16       v16bf;
typedef __attribute__((ext_vector_type(8)))  float        v8f;
typedef __attribute__((ext_vector_type(8)))  unsigned int v8u;

__device__ __forceinline__ unsigned short f2bf(float f) {
    unsigned int u = __float_as_uint(f);
    u = (u + 0x7FFFu + ((u >> 16) & 1u)) >> 16;     // round-to-nearest-even
    return (unsigned short)u;
}
__device__ __forceinline__ unsigned int pack2bf(float lo, float hi) {
    return (unsigned int)f2bf(lo) | ((unsigned int)f2bf(hi) << 16);
}

__device__ __forceinline__ v8f wmma_bf16(v8u a, v8u b, v8f c) {
    return __builtin_amdgcn_wmma_f32_16x16x32_bf16(
        false, __builtin_bit_cast(v16bf, a),
        false, __builtin_bit_cast(v16bf, b),
        (short)0, c, false, false);
}

// Build a v8u operand from two consecutive uint4 quads (2x b128 loads).
__device__ __forceinline__ v8u frag_from_quads(const uint4* p, int qidx) {
    uint4 a = p[qidx];
    uint4 b = p[qidx + 1];
    v8u r;
    r[0] = a.x; r[1] = a.y; r[2] = a.z; r[3] = a.w;
    r[4] = b.x; r[5] = b.y; r[6] = b.z; r[7] = b.w;
    return r;
}

// A-operand fragment (16-bit 16x32): lane's data is quads
// [2*hi] and [2*hi+1] of a 32-element (16-dword) row (k = 16*hi + s).
__device__ __forceinline__ v8u load_frag_a(const unsigned short* rowbase,
                                           int hi) {
    const uint4* p = (const uint4*)rowbase;
    uint4 a = p[2 * hi];
    uint4 b = p[2 * hi + 1];
    v8u r;
    r[0] = a.x; r[1] = a.y; r[2] = a.z; r[3] = a.w;
    r[4] = b.x; r[5] = b.y; r[6] = b.z; r[7] = b.w;
    return r;
}

// ---------------------------------------------------------------------------
// 16-lane-group XOR butterflies via v_permlane16_b32 (pure VALU).
// ---------------------------------------------------------------------------
__device__ constexpr unsigned selhalf(int k, int base) {
    unsigned s = 0;
    for (int i = 0; i < 8; i++)
        s |= (unsigned)(((base + i) ^ k) & 0xF) << (4 * i);
    return s;
}
template <int K>
__device__ __forceinline__ float lanexor16(float v) {
#if __has_builtin(__builtin_amdgcn_permlane16)
    constexpr unsigned slo = selhalf(K, 0);
    constexpr unsigned shi = selhalf(K, 8);
    return __uint_as_float(__builtin_amdgcn_permlane16(
        __float_as_uint(v), __float_as_uint(v), slo, shi, false, false));
#else
    return __shfl_xor(v, K, 32);
#endif
}
__device__ __forceinline__ float red_max16(float v) {
    v = fmaxf(v, lanexor16<8>(v));
    v = fmaxf(v, lanexor16<4>(v));
    v = fmaxf(v, lanexor16<2>(v));
    v = fmaxf(v, lanexor16<1>(v));
    return v;
}
__device__ __forceinline__ float red_sum16(float v) {
    v += lanexor16<8>(v);
    v += lanexor16<4>(v);
    v += lanexor16<2>(v);
    v += lanexor16<1>(v);
    return v;
}

// ---------------------------------------------------------------------------
// Async global->LDS staging (CDNA5 GLOBAL_LOAD_ASYNC_TO_LDS_*), ASYNCcnt.
// ---------------------------------------------------------------------------
#if __has_builtin(__builtin_amdgcn_global_load_async_to_lds_b128)
#define HAVE_ASYNC_LDS 1
#endif

#ifdef HAVE_ASYNC_LDS
typedef int async_v4i __attribute__((vector_size(16)));
typedef __attribute__((address_space(1))) async_v4i* g_v4i_p;
typedef __attribute__((address_space(3))) async_v4i* l_v4i_p;

__device__ __forceinline__ void async_copy_b128(const void* g, const void* l) {
    __builtin_amdgcn_global_load_async_to_lds_b128(
        (g_v4i_p)(unsigned long long)g,
        (l_v4i_p)(unsigned int)(unsigned long long)l, 0, 0);
}
__device__ __forceinline__ void wait_async0() {
#if __has_builtin(__builtin_amdgcn_s_wait_asynccnt)
    __builtin_amdgcn_s_wait_asynccnt(0);
#else
    asm volatile("s_wait_asynccnt 0x0" ::: "memory");
#endif
}
#else
__device__ __forceinline__ void async_copy_b128(const void* g, const void* l) {
    *(uint4*)l = *(const uint4*)g;
}
__device__ __forceinline__ void wait_async0() {}
#endif

// ---------------------------------------------------------------------------
// Weight prep: K-pair-packed u32 operands, laid out [n][K/2] (n-major) so
// GEMM B staging is a straight contiguous copy per n-row.
// ---------------------------------------------------------------------------
__device__ __forceinline__ float wb_val(int k, int n,
                                        const float* Wqkv, const float* Bm,
                                        const float* bb) {
    if (k < DIM) return Wqkv[(size_t)n * DIM + k];                 // Wqkv^T
    if (k < DIM + ER) {
        int er = k - DIM;
        return Bm[((size_t)(er >> 4) * QKV3 + n) * RANK + (er & 15)] * SCALING;
    }
    if (k < DIM + ER + NEXP)
        return bb[(size_t)(k - DIM - ER) * QKV3 + n] * SCALING;
    return 0.0f;                                                   // pad
}

__global__ void prep_wbp(const float* __restrict__ Wqkv,
                         const float* __restrict__ Bm,
                         const float* __restrict__ bb,
                         unsigned int* __restrict__ WBp) {
    int idx = blockIdx.x * 256 + threadIdx.x;
    if (idx >= QKV3 * (KCAT / 2)) return;
    int n = idx / (KCAT / 2), k2 = idx % (KCAT / 2);
    WBp[idx] = pack2bf(wb_val(2 * k2, n, Wqkv, Bm, bb),
                       wb_val(2 * k2 + 1, n, Wqkv, Bm, bb));
}

__global__ void prep_wpp(const float* __restrict__ Wproj,
                         unsigned int* __restrict__ WPp) {
    int idx = blockIdx.x * 256 + threadIdx.x;
    if (idx >= DIM * (DIM / 2)) return;
    int n = idx / (DIM / 2), k2 = idx % (DIM / 2);
    const float* src = Wproj + (size_t)n * DIM + 2 * k2;
    WPp[idx] = pack2bf(src[0], src[1]);
}

__global__ void prep_afp(const float* __restrict__ A,
                         unsigned int* __restrict__ Afp) {
    int idx = blockIdx.x * 256 + threadIdx.x;
    if (idx >= ER * (DIM / 2)) return;
    int er = idx / (DIM / 2), k2 = idx % (DIM / 2);
    const float* src = A + (size_t)er * DIM + 2 * k2;
    Afp[idx] = pack2bf(src[0], src[1]);
}

// ---------------------------------------------------------------------------
// Router: one wave per token. Softmax top-2 combine weights; writes bf16 x
// into XB[:, :768], combine into XB[:, 896:904], zeros into the K pad.
// ---------------------------------------------------------------------------
__global__ __launch_bounds__(256) void router_kernel(
    const float* __restrict__ x, const float* __restrict__ Wg,
    const float* __restrict__ bg, float* __restrict__ combine,
    unsigned short* __restrict__ XB) {
    const int wave = threadIdx.x >> 5, lane = threadIdx.x & 31;
    const int t = blockIdx.x * 8 + wave;
    const float* xp = x + (size_t)t * DIM;
    unsigned short* xb = XB + (size_t)t * KCAT;

    float p[NEXP];
#pragma unroll
    for (int e = 0; e < NEXP; e++) p[e] = 0.0f;
    for (int d = lane; d < DIM; d += 32) {
        float xv = xp[d];
        xb[d] = f2bf(xv);
#pragma unroll
        for (int e = 0; e < NEXP; e++) p[e] += xv * Wg[e * DIM + d];
    }
#pragma unroll
    for (int e = 0; e < NEXP; e++) {
        p[e] += __shfl_xor(p[e], 16, 32);
        p[e] = red_sum16(p[e]);
    }

    float lg[NEXP], mx = -1e30f;
#pragma unroll
    for (int e = 0; e < NEXP; e++) { lg[e] = p[e] + bg[e]; mx = fmaxf(mx, lg[e]); }
    float se = 0.0f;
#pragma unroll
    for (int e = 0; e < NEXP; e++) { lg[e] = __expf(lg[e] - mx); se += lg[e]; }
    float inv = 1.0f / se;
#pragma unroll
    for (int e = 0; e < NEXP; e++) lg[e] *= inv;

    int i1 = 0; float v1 = lg[0];
#pragma unroll
    for (int e = 1; e < NEXP; e++) if (lg[e] > v1) { v1 = lg[e]; i1 = e; }
    int i2 = -1; float v2 = -1.0f;
#pragma unroll
    for (int e = 0; e < NEXP; e++)
        if (e != i1 && lg[e] > v2) { v2 = lg[e]; i2 = e; }

    if (lane == 0) {
#pragma unroll
        for (int e = 0; e < NEXP; e++) {
            float c = (e == i1) ? v1 : ((e == i2) ? v2 : 0.0f);
            combine[t * NEXP + e] = c;
            xb[DIM + ER + e] = f2bf(c);
        }
    }
    if (lane < 24) xb[DIM + ER + NEXP + lane] = 0;   // zero K pad
}

// ---------------------------------------------------------------------------
// WMMA GEMM:  C[M,N] = A[M,K](bf16 row-major, stride lda)
//                    * Bp[N][K/2](K-pair-packed u32, row stride ldb) + bias
// Block = 256 threads (8 waves), tile 128x128, K-step 32, double-buffered
// async global->LDS staging. Wave computes 32x64 = 2x4 C fragments.
// MODE 0: (+ba, *combine[t,e]) -> bf16   (LoRA h, N=128)
// MODE 1: +bias, Q cols *ATT_SCALE -> bf16 qkv   (N=2304)
// MODE 2: +bias -> f32 out               (N=768)
// ---------------------------------------------------------------------------
template <int MODE>
__global__ __launch_bounds__(256) void gemm_bf16_kernel(
    const unsigned short* __restrict__ A, int lda,
    const unsigned int* __restrict__ Bp, int N, int K, int ldb,
    const float* __restrict__ bias, const float* __restrict__ combine,
    unsigned short* __restrict__ outb, float* __restrict__ outf, int ldo) {
    const int tid = threadIdx.x;
    const int wave = tid >> 5, lane = tid & 31;
    const int ln15 = lane & 15, hi = lane >> 4;
    const int nt = blockIdx.x, mt = blockIdx.y;
    const int wm = (wave & 3) * 32, wn = (wave >> 2) * 64;

    __shared__ __align__(16) unsigned short Alds[2][128 * A_STRIDE];
    __shared__ __align__(16) unsigned int   Blds[2][128 * B_STRIDE];

    v8f acc[2][4];
#pragma unroll
    for (int mi = 0; mi < 2; mi++)
#pragma unroll
        for (int ni = 0; ni < 4; ni++)
#pragma unroll
            for (int j = 0; j < 8; j++) acc[mi][ni][j] = 0.0f;

    const int arow = tid >> 1, acol = (tid & 1) * 16;  // A: 16 bf16/thread
    const int bn = tid >> 2, bq = tid & 3;             // B: 2 quads/thread

    auto stage = [&](int buf, int k0) {
        const unsigned short* ap =
            A + (size_t)(mt * 128 + arow) * lda + k0 + acol;
        unsigned short* ad = &Alds[buf][arow * A_STRIDE + acol];
        async_copy_b128(ap, ad);
        async_copy_b128(ap + 8, ad + 8);
        const unsigned int* bbase =
            Bp + (size_t)(nt * 128) * ldb + (k0 >> 1) + bq * 4;
        async_copy_b128(bbase + (size_t)bn * ldb,
                        &Blds[buf][bn * B_STRIDE + bq * 4]);
        async_copy_b128(bbase + (size_t)(bn + 64) * ldb,
                        &Blds[buf][(bn + 64) * B_STRIDE + bq * 4]);
    };

    const int nK = K / 32;
    stage(0, 0);
    wait_async0();
    __syncthreads();

    for (int kt = 0; kt < nK; kt++) {
        const int cur = kt & 1;
        if (kt + 1 < nK) stage(1 - cur, (kt + 1) * 32);   // prefetch next tile

        v8u af[2], bf_[4];
#pragma unroll
        for (int mi = 0; mi < 2; mi++)
            af[mi] = load_frag_a(
                &Alds[cur][(wm + mi * 16 + ln15) * A_STRIDE], hi);
#pragma unroll
        for (int ni = 0; ni < 4; ni++)
            bf_[ni] = frag_from_quads(
                (const uint4*)&Blds[cur][(wn + ni * 16 + ln15) * B_STRIDE],
                hi * 2);
#pragma unroll
        for (int mi = 0; mi < 2; mi++)
#pragma unroll
            for (int ni = 0; ni < 4; ni++)
                acc[mi][ni] = wmma_bf16(af[mi], bf_[ni], acc[mi][ni]);

        if (kt + 1 < nK) {
            wait_async0();     // next tile landed
            __syncthreads();   // and everyone is done reading current tile
        }
    }

    // epilogue
#pragma unroll
    for (int mi = 0; mi < 2; mi++)
#pragma unroll
        for (int ni = 0; ni < 4; ni++) {
            const int n = nt * 128 + wn + ni * 16 + ln15;
            const float bv = bias[n];
#pragma unroll
            for (int j = 0; j < 8; j++) {
                const int row = mt * 128 + wm + mi * 16 + j + hi * 8;
                float v = acc[mi][ni][j] + bv;
                if (MODE == 0) {
                    v *= combine[row * NEXP + (n >> 4)];
                    outb[(size_t)row * ldo + n] = f2bf(v);
                } else if (MODE == 1) {
                    if (n < DIM) v *= ATT_SCALE;   // pre-scale Q columns
                    outb[(size_t)row * ldo + n] = f2bf(v);
                } else {
                    outf[(size_t)row * ldo + n] = v;
                }
            }
        }
}

// ---------------------------------------------------------------------------
// Flash attention: 4 waves/block, each wave owns 16 query rows; 96 (b,h)
// pairs x 16 query blocks. QK^T and P*V via bf16 WMMA, online softmax (f32).
// K tile staged async as [key][e/2] (straight copy); V tile as [e][key/2].
// ---------------------------------------------------------------------------
__global__ __launch_bounds__(128) void attn_kernel(
    const unsigned short* __restrict__ qkv, unsigned short* __restrict__ ao) {
    const int bh = blockIdx.x >> 4, qblk = blockIdx.x & 15;
    const int b = bh / NHEAD, h = bh % NHEAD;
    const int wave = threadIdx.x >> 5, lane = threadIdx.x & 31;
    const int ln15 = lane & 15, hi = lane >> 4;
    const int tbase = b * 1024;
    const int kcol = DIM + h * HD, vcol = 2 * DIM + h * HD;

    __shared__ __align__(16) unsigned int   Kt[32 * K_STRIDE]; // [key][e2]
    __shared__ __align__(16) unsigned int   Vt[64 * V_STRIDE]; // [e][key2]
    __shared__ __align__(16) unsigned short Pst[4][16 * 32];

    // Q fragments (two K=32 slabs covering head dim 64); Q is pre-scaled.
    v8u qf[2];
    {
        const int sq = qblk * 64 + wave * 16 + ln15;
        const unsigned short* qp = qkv + (size_t)(tbase + sq) * QKV3 + h * HD;
#pragma unroll
        for (int kk = 0; kk < 2; kk++)
            qf[kk] = load_frag_a(qp + kk * 32, hi);
    }

    v8f o[4];
    float m[8], lsum[8];
#pragma unroll
    for (int f = 0; f < 4; f++)
#pragma unroll
        for (int j = 0; j < 8; j++) o[f][j] = 0.0f;
#pragma unroll
    for (int j = 0; j < 8; j++) { m[j] = -1e30f; lsum[j] = 0.0f; }

    for (int kt = 0; kt < 32; kt++) {
        __syncthreads();
        // K tile: 32 keys x 64 dims -> Kt[key][e2], straight async copy
#pragma unroll
        for (int r = 0; r < 2; r++) {
            int idx = r * 128 + threadIdx.x;         // 256 b128 chunks
            int key = idx >> 3, q = idx & 7;
            async_copy_b128(
                qkv + (size_t)(tbase + kt * 32 + key) * QKV3 + kcol + q * 8,
                &Kt[key * K_STRIDE + q * 4]);
        }
        // V tile: packed key pairs -> Vt[e][key2] (interleave, sync)
#pragma unroll
        for (int r = 0; r < 4; r++) {
            int idx = r * 128 + threadIdx.x;
            int ep = idx & 31, key2 = idx >> 5;
            const size_t rowa = (size_t)(tbase + kt * 32 + 2 * key2) * QKV3;
            unsigned int a = *(const unsigned int*)(qkv + rowa + vcol + 2 * ep);
            unsigned int c =
                *(const unsigned int*)(qkv + rowa + QKV3 + vcol + 2 * ep);
            Vt[(2 * ep) * V_STRIDE + key2]     = (a & 0xFFFFu) | (c << 16);
            Vt[(2 * ep + 1) * V_STRIDE + key2] = (a >> 16) | (c & 0xFFFF0000u);
        }
        wait_async0();
        __syncthreads();

        // S = Q K^T  (16 x 32 scores per wave)
        v8f sc[2];
#pragma unroll
        for (int nh = 0; nh < 2; nh++) {
            v8f c;
#pragma unroll
            for (int j = 0; j < 8; j++) c[j] = 0.0f;
            const uint4* kp =
                (const uint4*)&Kt[(nh * 16 + ln15) * K_STRIDE];
#pragma unroll
            for (int kk = 0; kk < 2; kk++)
                c = wmma_bf16(qf[kk], frag_from_quads(kp, kk * 4 + hi * 2), c);
            sc[nh] = c;
        }

        // online softmax per row (row = j + hi*8); scores already scaled
        float alpha[8];
#pragma unroll
        for (int j = 0; j < 8; j++) {
            float s0 = sc[0][j], s1 = sc[1][j];
            float mx = red_max16(fmaxf(s0, s1));
            float mn = fmaxf(m[j], mx);
            alpha[j] = __expf(m[j] - mn);
            m[j] = mn;
            float p0 = __expf(s0 - mn), p1 = __expf(s1 - mn);
            sc[0][j] = p0; sc[1][j] = p1;
            lsum[j] = lsum[j] * alpha[j] + red_sum16(p0 + p1);
        }
#pragma unroll
        for (int f = 0; f < 4; f++)
#pragma unroll
            for (int j = 0; j < 8; j++) o[f][j] *= alpha[j];

        // P (bf16) -> LDS, reload as A fragment, O += P * V
        unsigned short* pw = &Pst[wave][0];
#pragma unroll
        for (int j = 0; j < 8; j++) {
            int row = j + hi * 8;
            pw[row * 32 + ln15]      = f2bf(sc[0][j]);
            pw[row * 32 + 16 + ln15] = f2bf(sc[1][j]);
        }
        __syncthreads();
        v8u pf = load_frag_a(pw + ln15 * 32, hi);
#pragma unroll
        for (int f = 0; f < 4; f++) {
            const uint4* vp =
                (const uint4*)&Vt[(f * 16 + ln15) * V_STRIDE];
            o[f] = wmma_bf16(pf, frag_from_quads(vp, hi * 2), o[f]);
        }
    }

    // finalize: O / l, store bf16 [t][h*64+e]
    const int sq0 = qblk * 64 + wave * 16;
#pragma unroll
    for (int f = 0; f < 4; f++)
#pragma unroll
        for (int j = 0; j < 8; j++) {
            int row = sq0 + j + hi * 8;
            float v = o[f][j] / lsum[j];
            ao[(size_t)(tbase + row) * DIM + h * HD + f * 16 + ln15] = f2bf(v);
        }
}

// ---------------------------------------------------------------------------
// Host launcher
// ---------------------------------------------------------------------------
extern "C" void kernel_launch(void* const* d_in, const int* in_sizes, int n_in,
                              void* d_out, int out_size, void* d_ws,
                              size_t ws_size, hipStream_t stream) {
    const float* x     = (const float*)d_in[0];
    const float* Wg    = (const float*)d_in[1];
    const float* bg    = (const float*)d_in[2];
    const float* Wqkv  = (const float*)d_in[3];
    const float* bqkv  = (const float*)d_in[4];
    const float* Wproj = (const float*)d_in[5];
    const float* bproj = (const float*)d_in[6];
    const float* A     = (const float*)d_in[7];
    const float* ba    = (const float*)d_in[8];
    const float* Bm    = (const float*)d_in[9];
    const float* bb    = (const float*)d_in[10];

    char* ws = (char*)d_ws;
    size_t off = 0;
    auto carve = [&](size_t bytes) {
        void* p = ws + off;
        off += (bytes + 255) & ~(size_t)255;
        return p;
    };
    unsigned int*   WBp = (unsigned int*)carve((size_t)QKV3 * (KCAT / 2) * 4);
    unsigned int*   WPp = (unsigned int*)carve((size_t)DIM * (DIM / 2) * 4);
    unsigned int*   Afp = (unsigned int*)carve((size_t)ER * (DIM / 2) * 4);
    float*          cmb = (float*)carve((size_t)T_TOK * NEXP * 4);
    unsigned short* XB  = (unsigned short*)carve((size_t)T_TOK * KCAT * 2);
    unsigned short* QKV = (unsigned short*)carve((size_t)T_TOK * QKV3 * 2);
    unsigned short* AO  = (unsigned short*)carve((size_t)T_TOK * DIM * 2);

    prep_wbp<<<(QKV3 * (KCAT / 2) + 255) / 256, 256, 0, stream>>>(
        Wqkv, Bm, bb, WBp);
    prep_wpp<<<(DIM * (DIM / 2) + 255) / 256, 256, 0, stream>>>(Wproj, WPp);
    prep_afp<<<(ER * (DIM / 2) + 255) / 256, 256, 0, stream>>>(A, Afp);

    router_kernel<<<T_TOK / 8, 256, 0, stream>>>(x, Wg, bg, cmb, XB);

    // LoRA h: [T,768] x [768,128] -> XB[:,768:896] (bf16, *combine, +ba)
    gemm_bf16_kernel<0><<<dim3(1, T_TOK / 128), 256, 0, stream>>>(
        XB, KCAT, Afp, ER, DIM, DIM / 2, ba, cmb, XB + DIM, nullptr, KCAT);

    // fused qkv + LoRA delta: [T,928] x [928,2304] + bqkv -> bf16 qkv
    gemm_bf16_kernel<1><<<dim3(QKV3 / 128, T_TOK / 128), 256, 0, stream>>>(
        XB, KCAT, WBp, QKV3, KCAT, KCAT / 2, bqkv, nullptr, QKV, nullptr,
        QKV3);

    // flash attention over 96 (b,h) pairs x 16 query blocks
    attn_kernel<<<96 * 16, 128, 0, stream>>>(QKV, AO);

    // output projection: [T,768] x [768,768] + bproj -> f32 out
    gemm_bf16_kernel<2><<<dim3(DIM / 128, T_TOK / 128), 256, 0, stream>>>(
        AO, DIM, WPp, DIM, DIM, DIM / 2, bproj, nullptr, nullptr,
        (float*)d_out, DIM);
}
